// SelfAttention_50912542326895
// MI455X (gfx1250) — matrix operand order repaired and checked by reading, not compiled, not run on hardware
//
#include <hip/hip_runtime.h>
#include <hip/hip_bf16.h>

// ---------------------------------------------------------------------------
// Self-attention (B=2, S=2048, E=1024, H=16, D=64), causal mask, softmax
// scaled by 1/sqrt(E). bf16 WMMA compute with f32 accumulation.
// ---------------------------------------------------------------------------

#define B_  2
#define S_  2048
#define E_  1024
#define H_  16
#define D_  64
#define MTOT (B_ * S_)          // 4096

typedef __attribute__((ext_vector_type(16))) __bf16 v16bf;
typedef __attribute__((ext_vector_type(8)))  __bf16 v8bf;
typedef __attribute__((ext_vector_type(8)))  float  v8f;
typedef int v4i_gcc __attribute__((vector_size(16)));   // matches builtin pointee

// ---- CDNA5 async global->LDS copy (guarded; falls back to vector copy) ----
#if defined(__has_builtin)
# if __has_builtin(__builtin_amdgcn_global_load_async_to_lds_b128)
#  define HAVE_ASYNC_LDS 1
# endif
# if __has_builtin(__builtin_amdgcn_s_wait_asynccnt)
#  define WAIT_ASYNC() __builtin_amdgcn_s_wait_asynccnt(0)
# endif
#endif
#ifndef WAIT_ASYNC
# define WAIT_ASYNC() asm volatile("s_wait_asynccnt 0x0" ::: "memory")
#endif

__device__ __forceinline__ v16bf join16(v8bf lo, v8bf hi) {
    v16bf r;
#pragma unroll
    for (int i = 0; i < 8; ++i) { r[i] = lo[i]; r[i + 8] = hi[i]; }
    return r;
}

__device__ __forceinline__ unsigned pack2(float a, float b) {
    union { __bf16 h[2]; unsigned u; } x;
    x.h[0] = (__bf16)a; x.h[1] = (__bf16)b;
    return x.u;
}

// ---------------------------------------------------------------------------
// Weight convert + transpose: Wt[k][n] = (bf16) W[n][k]
// ---------------------------------------------------------------------------
__global__ __launch_bounds__(256)
void wt_convert_kernel(const float* __restrict__ W, unsigned short* __restrict__ Wt_u) {
    __bf16* Wt = (__bf16*)Wt_u;
    int idx = blockIdx.x * 256 + threadIdx.x;   // over E*E, coalesced read
    int n = idx >> 10;
    int k = idx & (E_ - 1);
    Wt[(size_t)k * E_ + n] = (__bf16)W[idx];
}

// ---------------------------------------------------------------------------
// GEMM: C[M=4096, N=1024] = X[M, K=1024] @ Wt[K, N] + bias[N]
//  IN_BF16: X is bf16 (else f32)
//  OUTMODE 0: bf16 head-split  out[b][h][s][d]
//  OUTMODE 1: bf16 K-transpose out[b][h][d][s]
//  OUTMODE 2: f32 row-major    out[m][n]
// Block 256 thr (8 waves), tile 128x64, BK=32, wave tile 32x32 (2x2 WMMA).
// W tile staged via GLOBAL_LOAD_ASYNC_TO_LDS_B128 when available.
// ---------------------------------------------------------------------------
template <bool IN_BF16, int OUTMODE>
__global__ __launch_bounds__(256)
void gemm_kernel(const void* __restrict__ xv,
                 const unsigned short* __restrict__ Wt_u,
                 const float* __restrict__ bias,
                 void* __restrict__ outv) {
    constexpr int BM = 128, BN = 64, BK = 32;
    constexpr int N = E_, K = E_;
    __shared__ __bf16 Xs[BM][BK + 8];   // 40 halves/row = 80B (16B aligned, bank-spread)
    __shared__ __bf16 Ws[BK][BN + 8];   // 72 halves/row = 144B

    const __bf16* Wt = (const __bf16*)Wt_u;
    const float*  Xf = (const float*)xv;
    const __bf16* Xb = (const __bf16*)xv;

    const int tid  = threadIdx.x;
    const int wave = tid >> 5, lane = tid & 31;
    const int wr = wave >> 1, wc = wave & 1;          // 4x2 wave grid
    const int lm = lane & 15, lh = lane >> 4;
    const int m0 = blockIdx.y * BM;
    const int n0 = blockIdx.x * BN;

    // staging assignments
    const int sr = tid >> 1;            // X row 0..127
    const int sc = (tid & 1) * 16;      // X col 0/16
    const int wrow = tid >> 3;          // W row 0..31
    const int wcol = (tid & 7) * 8;     // W col 0..56

    v8f acc[2][2] = {};

    for (int k0 = 0; k0 < K; k0 += BK) {
        // ---- stage Wt tile (async DMA to LDS when the builtin exists) ----
        {
            const __bf16* src = Wt + (size_t)(k0 + wrow) * N + n0 + wcol;
#if defined(HAVE_ASYNC_LDS)
            __builtin_amdgcn_global_load_async_to_lds_b128(
                (__attribute__((address_space(1))) v4i_gcc*)(unsigned long long)src,
                (__attribute__((address_space(3))) v4i_gcc*)
                    (unsigned)(unsigned long long)&Ws[wrow][wcol],
                0, 0);
#else
            *(uint4*)&Ws[wrow][wcol] = *(const uint4*)src;
#endif
        }
        // ---- stage X tile (convert f32 -> bf16 when needed) ----
        if (IN_BF16) {
            const __bf16* src = Xb + (size_t)(m0 + sr) * K + k0 + sc;
            *(uint4*)&Xs[sr][sc] = *(const uint4*)src;
        } else {
            const float4* s4 = (const float4*)(Xf + (size_t)(m0 + sr) * K + k0 + sc);
            float4 f0 = s4[0], f1 = s4[1], f2 = s4[2], f3 = s4[3];
            unsigned* dp = (unsigned*)&Xs[sr][sc];
            dp[0] = pack2(f0.x, f0.y); dp[1] = pack2(f0.z, f0.w);
            dp[2] = pack2(f1.x, f1.y); dp[3] = pack2(f1.z, f1.w);
            dp[4] = pack2(f2.x, f2.y); dp[5] = pack2(f2.z, f2.w);
            dp[6] = pack2(f3.x, f3.y); dp[7] = pack2(f3.z, f3.w);
        }
        // prefetch next K-step sources into cache
        if (k0 + BK < K) {
            if (IN_BF16)
                __builtin_prefetch(Xb + (size_t)(m0 + sr) * K + k0 + BK + sc, 0, 3);
            else
                __builtin_prefetch(Xf + (size_t)(m0 + sr) * K + k0 + BK + sc, 0, 3);
            __builtin_prefetch(Wt + (size_t)(k0 + BK + wrow) * N + n0 + wcol, 0, 3);
        }
#if defined(HAVE_ASYNC_LDS)
        WAIT_ASYNC();
#endif
        __syncthreads();

        // ---- fragments + WMMA ----
        v16bf a[2], bf[2];
#pragma unroll
        for (int mt = 0; mt < 2; ++mt) {
            const __bf16* base = &Xs[wr * 32 + mt * 16 + lm][lh * 8];
            v8bf lo = *(const v8bf*)base;
            v8bf hi = *(const v8bf*)(base + 16);
            a[mt] = join16(lo, hi);
        }
#pragma unroll
        for (int nt = 0; nt < 2; ++nt) {
#pragma unroll
            for (int s = 0; s < 16; ++s)
                bf[nt][s] = Ws[s < 8 ? lh * 8 + s : 16 + lh * 8 + (s - 8)][wc * 32 + nt * 16 + lm];
        }
#pragma unroll
        for (int mt = 0; mt < 2; ++mt)
#pragma unroll
            for (int nt = 0; nt < 2; ++nt)
                acc[mt][nt] = __builtin_amdgcn_wmma_f32_16x16x32_bf16(
                    false, a[mt], false, bf[nt], (short)0, acc[mt][nt], false, false);
        __syncthreads();
    }

    // ---- epilogue: bias + layout-specific store ----
#pragma unroll
    for (int mt = 0; mt < 2; ++mt)
#pragma unroll
        for (int nt = 0; nt < 2; ++nt)
#pragma unroll
            for (int r = 0; r < 8; ++r) {
                int m = m0 + wr * 32 + mt * 16 + lh * 8 + r;
                int n = n0 + wc * 32 + nt * 16 + lm;
                float v = acc[mt][nt][r] + bias[n];
                if constexpr (OUTMODE == 2) {
                    ((float*)outv)[(size_t)m * N + n] = v;
                } else {
                    int bb = m >> 11;           // m / S
                    int s  = m & (S_ - 1);
                    int h  = n >> 6;            // n / 64
                    int d  = n & 63;
                    __bf16* o = (__bf16*)outv;
                    if constexpr (OUTMODE == 0)
                        o[(((size_t)bb * H_ + h) * S_ + s) * D_ + d] = (__bf16)v;
                    else
                        o[(((size_t)bb * H_ + h) * D_ + d) * S_ + s] = (__bf16)v;
                }
            }
}

// ---------------------------------------------------------------------------
// Flash-style causal attention per (b, h, 64-row q tile).
// 4 waves x 16 q-rows. Online softmax; 8 WMMAs per 32-key block; causal
// blocks beyond the diagonal are skipped entirely. K/V fragments for the
// whole key block are issued up front so V loads overlap the softmax VALU.
// ---------------------------------------------------------------------------
__global__ __launch_bounds__(128)
void attn_kernel(const unsigned short* __restrict__ qh_u,
                 const unsigned short* __restrict__ kT_u,
                 const unsigned short* __restrict__ vh_u,
                 unsigned short* __restrict__ ao_u) {
    __shared__ __bf16 Ps[4][16][40];    // per-wave probs tile, 80B rows

    const __bf16* qh = (const __bf16*)qh_u;
    const __bf16* kT = (const __bf16*)kT_u;
    const __bf16* vh = (const __bf16*)vh_u;
    __bf16* ao = (__bf16*)ao_u;

    const int tid  = threadIdx.x;
    const int wave = tid >> 5, lane = tid & 31;
    const int lm = lane & 15, lh = lane >> 4;
    const int qt = blockIdx.x, h = blockIdx.y, bb = blockIdx.z;
    const int q0 = qt * 64 + wave * 16;

    const size_t headQ = ((size_t)bb * H_ + h) * S_ * D_;   // qh / vh base
    const size_t headK = ((size_t)bb * H_ + h) * D_ * S_;   // kT base

    // Q fragments (two 16x32 d-chunks), reused across all key blocks
    v16bf qf[2];
#pragma unroll
    for (int c = 0; c < 2; ++c) {
        const __bf16* base = qh + headQ + (size_t)(q0 + lm) * D_ + c * 32 + lh * 8;
        v8bf lo = *(const v8bf*)base;
        v8bf hi = *(const v8bf*)(base + 16);
        qf[c] = join16(lo, hi);
    }

    v8f o[4] = {};
    float mrow[8], lrow[8];
#pragma unroll
    for (int r = 0; r < 8; ++r) { mrow[r] = -3.0e38f; lrow[r] = 0.f; }

    const float SCALE = 0.03125f;               // 1/sqrt(1024)
    const float L2E   = 1.44269504088896341f;

    const int kbEnd = qt * 2 + 2;               // causal: keys <= q0+63 only
    for (int kb = 0; kb < kbEnd; ++kb) {
        const int kbase = kb * 32;

        // prefetch next block's K/V rows
        if (kb + 1 < kbEnd) {
            __builtin_prefetch(kT + headK + (size_t)lane * S_ + kbase + 32, 0, 3);
            __builtin_prefetch(vh + headQ + (size_t)(kbase + 32 + lane) * D_, 0, 3);
        }

        // ---- issue ALL K^T and V fragment loads for this block up front ----
        v16bf kf[2][2];                          // [d-chunk][key tile]
#pragma unroll
        for (int c = 0; c < 2; ++c) {
#pragma unroll
            for (int t = 0; t < 2; ++t)
#pragma unroll
                for (int s = 0; s < 16; ++s)
                    kf[c][t][s] = kT[headK + (size_t)(c * 32 + (s < 8 ? lh * 8 + s : 16 + lh * 8 + (s - 8))) * S_ + kbase + t * 16 + lm];
        }
        v16bf vf[4];                             // [d tile], used after softmax
        {
#pragma unroll
            for (int nt = 0; nt < 4; ++nt)
#pragma unroll
                for (int s = 0; s < 16; ++s)
                    vf[nt][s] = vh[headQ + (size_t)(kbase + (s < 8 ? lh * 8 + s : 16 + lh * 8 + (s - 8))) * D_ + nt * 16 + lm];
        }

        // ---- E = Q K^T (two 16-key column tiles, K-dim = D in 2 steps) ----
        v8f e[2] = {};
#pragma unroll
        for (int c = 0; c < 2; ++c)
#pragma unroll
            for (int t = 0; t < 2; ++t)
                e[t] = __builtin_amdgcn_wmma_f32_16x16x32_bf16(
                    false, qf[c], false, kf[c][t], (short)0, e[t], false, false);

        // ---- scale + causal mask + online softmax ----
        float p[2][8], xm[8];
#pragma unroll
        for (int r = 0; r < 8; ++r) {
            int qg = q0 + lh * 8 + r;
            float x0 = (kbase + lm      <= qg) ? e[0][r] * SCALE : -3.0e38f;
            float x1 = (kbase + 16 + lm <= qg) ? e[1][r] * SCALE : -3.0e38f;
            p[0][r] = x0; p[1][r] = x1;
            xm[r] = fmaxf(x0, x1);
        }
#pragma unroll
        for (int off = 8; off >= 1; off >>= 1)
#pragma unroll
            for (int r = 0; r < 8; ++r)
                xm[r] = fmaxf(xm[r], __shfl_xor(xm[r], off));

        float corr[8];
#pragma unroll
        for (int r = 0; r < 8; ++r) {
            float mn = fmaxf(mrow[r], xm[r]);
            corr[r]  = __builtin_exp2f((mrow[r] - mn) * L2E);
            mrow[r]  = mn;
            p[0][r]  = __builtin_exp2f((p[0][r] - mn) * L2E);
            p[1][r]  = __builtin_exp2f((p[1][r] - mn) * L2E);
        }
        float rs[8];
#pragma unroll
        for (int r = 0; r < 8; ++r) rs[r] = p[0][r] + p[1][r];
#pragma unroll
        for (int off = 8; off >= 1; off >>= 1)
#pragma unroll
            for (int r = 0; r < 8; ++r) rs[r] += __shfl_xor(rs[r], off);
#pragma unroll
        for (int r = 0; r < 8; ++r) lrow[r] = lrow[r] * corr[r] + rs[r];
#pragma unroll
        for (int nt = 0; nt < 4; ++nt)
#pragma unroll
            for (int r = 0; r < 8; ++r) o[nt][r] *= corr[r];

        // ---- probs -> LDS (reshape C-layout to A-fragment layout) ----
#pragma unroll
        for (int t = 0; t < 2; ++t)
#pragma unroll
            for (int r = 0; r < 8; ++r)
                Ps[wave][lh * 8 + r][t * 16 + lm] = (__bf16)p[t][r];
        // same-wave LDS is in-order: store->load RAW is safe

        v16bf pf;
        {
            const __bf16* base = &Ps[wave][lm][lh * 8];
            v8bf lo = *(const v8bf*)base;
            v8bf hi = *(const v8bf*)(base + 16);
            pf = join16(lo, hi);
        }

        // ---- O += P V (four 16-wide d tiles) ----
#pragma unroll
        for (int nt = 0; nt < 4; ++nt)
            o[nt] = __builtin_amdgcn_wmma_f32_16x16x32_bf16(
                false, pf, false, vf[nt], (short)0, o[nt], false, false);
    }

    // ---- epilogue: normalize, store bf16 [B][S][E] ----
#pragma unroll
    for (int nt = 0; nt < 4; ++nt)
#pragma unroll
        for (int r = 0; r < 8; ++r) {
            float v = o[nt][r] / lrow[r];
            int s = q0 + lh * 8 + r;
            int d = nt * 16 + lm;
            ao[((size_t)bb * S_ + s) * E_ + h * D_ + d] = (__bf16)v;
        }
}

// ---------------------------------------------------------------------------
// Host launcher
// ---------------------------------------------------------------------------
extern "C" void kernel_launch(void* const* d_in, const int* in_sizes, int n_in,
                              void* d_out, int out_size, void* d_ws, size_t ws_size,
                              hipStream_t stream) {
    (void)in_sizes; (void)n_in; (void)out_size; (void)ws_size;
    const float* values = (const float*)d_in[0];
    const float* keys   = (const float*)d_in[1];
    const float* query  = (const float*)d_in[2];
    // d_in[3] = mask: causal tril by construction; applied analytically.
    const float* Wv = (const float*)d_in[4];
    const float* bv = (const float*)d_in[5];
    const float* Wk = (const float*)d_in[6];
    const float* bk = (const float*)d_in[7];
    const float* Wq = (const float*)d_in[8];
    const float* bq = (const float*)d_in[9];
    const float* Wo = (const float*)d_in[10];
    const float* bo = (const float*)d_in[11];

    const size_t WE  = (size_t)E_ * E_;       // 1M elems
    const size_t ACT = (size_t)B_ * S_ * E_;  // 4M elems
    unsigned short* WtQ = (unsigned short*)d_ws;
    unsigned short* WtK = WtQ + WE;
    unsigned short* WtV = WtK + WE;
    unsigned short* WtO = WtV + WE;
    unsigned short* qh  = WtO + WE;           // [B][H][S][D] bf16
    unsigned short* kT  = qh + ACT;           // [B][H][D][S] bf16
    unsigned short* vh  = kT + ACT;           // [B][H][S][D] bf16
    unsigned short* ao  = vh + ACT;           // [B][S][E]    bf16

    dim3 cgrid((unsigned)(WE / 256));
    wt_convert_kernel<<<cgrid, 256, 0, stream>>>(Wq, WtQ);
    wt_convert_kernel<<<cgrid, 256, 0, stream>>>(Wk, WtK);
    wt_convert_kernel<<<cgrid, 256, 0, stream>>>(Wv, WtV);
    wt_convert_kernel<<<cgrid, 256, 0, stream>>>(Wo, WtO);

    dim3 ggrid(E_ / 64, MTOT / 128);
    gemm_kernel<false, 0><<<ggrid, 256, 0, stream>>>(query,  WtQ, bq, qh);
    gemm_kernel<false, 1><<<ggrid, 256, 0, stream>>>(keys,   WtK, bk, kT);
    gemm_kernel<false, 0><<<ggrid, 256, 0, stream>>>(values, WtV, bv, vh);

    attn_kernel<<<dim3(S_ / 64, H_, B_), 128, 0, stream>>>(qh, kT, vh, ao);

    gemm_kernel<true, 2><<<ggrid, 256, 0, stream>>>(ao, WtO, bo, d_out);
}
